// InstanceSegmentationHead_5420248728321
// MI455X (gfx1250) — matrix operand order repaired and checked by reading, not compiled, not run on hardware
//
#include <hip/hip_runtime.h>
#include <hip/hip_bf16.h>
#include <math.h>

// ---------------------------------------------------------------------------
// InstanceSegmentationHead for gfx1250 (MI455X).
// All dense contractions (1x1 convs, ConvTranspose sub-GEMMs, 3x3 convs via
// 9-tap accumulation, prototype einsum) run through v_wmma_f32_16x16x32_f16,
// 4 independent accumulators per wave (16 oc x 64 pixels) to hide WMMA
// latency and reuse the weight fragment. Grouped per-prototype heads are
// fused in LDS (VALU). Requires ws_size >= ~322 MB.
// ---------------------------------------------------------------------------

typedef __attribute__((ext_vector_type(16))) _Float16 v16h;
typedef __attribute__((ext_vector_type(8)))  float    v8f;

#define HW96   9216
#define HW336  112896
#define W336   336

union AHalf { v16h v; _Float16 e[16]; };
union CAcc  { v8f  v; float    e[8];  };

// --------------------------- generic WMMA GEMM -----------------------------
// D[m][col(n)] = scale * sum_k A[m*sAm + k*sAk] * B[k*ldb + n]  (+bias/relu)
// col(n) = ((n/inW)*stride + aoff)*outW + (n%inW)*stride + boff
// (direct store: inW = huge, stride=1, outW=0, aoff=boff=0 -> col = n)
// Each wave: 16 (M) x 64 (N) tile, 4 accumulators sharing one A fragment.
__global__ void gemm16_wmma(const float* __restrict__ A, int sAm, int sAk,
                            const float* __restrict__ B, int ldb,
                            const float* __restrict__ bias, float scale, int relu,
                            float* __restrict__ D, int dplane, int K,
                            int inW, int outW, int stride, int aoff, int boff)
{
    const int l  = threadIdx.x;          // 0..31 (one wave)
    const int n0 = blockIdx.x * 64;
    const int m0 = blockIdx.y * 16;
    const int mA   = m0 + (l & 15);      // A: lane&15 selects row M
    const int khiA = (l >> 4) * 8;       // lanes 16..31 hold K+8 block

    AHalf a, b[4];
    CAcc  c[4];
#pragma unroll
    for (int t = 0; t < 4; ++t)
#pragma unroll
        for (int r = 0; r < 8; ++r) c[t].e[r] = 0.0f;

    for (int k0 = 0; k0 < K; k0 += 32) {
        // A fragment: 16x32 f16, lane=M, half h -> K = (h&7) + (h>=8)*16 + (lane>=16)*8
#pragma unroll
        for (int h = 0; h < 16; ++h) {
            int kk = k0 + (h & 7) + ((h >> 3) << 4) + khiA;
            a.e[h] = (_Float16)A[(size_t)mA * sAm + (size_t)kk * sAk];
        }
        // B fragments: 32x16 f16 each; lane&15 -> N = n0+t*16+(lane&15), half h -> K same as A
        const float* Bp = B + (size_t)k0 * ldb + n0 + (l & 15);
        if (k0 + 32 < K) __builtin_prefetch(Bp + 32 * (size_t)ldb, 0, 1);
#pragma unroll
        for (int t = 0; t < 4; ++t)
#pragma unroll
            for (int h = 0; h < 16; ++h) b[t].e[h] = (_Float16)Bp[(size_t)((h & 7) + ((h >> 3) << 4) + khiA) * ldb + t * 16];

#pragma unroll
        for (int t = 0; t < 4; ++t)
            c[t].v = __builtin_amdgcn_wmma_f32_16x16x32_f16(false, a.v, false, b[t].v,
                                                            (short)0, c[t].v, false, false);
    }

#pragma unroll
    for (int t = 0; t < 4; ++t) {
        const int n   = n0 + t * 16 + (l & 15);
        const int col = ((n / inW) * stride + aoff) * outW + (n % inW) * stride + boff;
#pragma unroll
        for (int r = 0; r < 8; ++r) {
            int   m = m0 + r + ((l >> 4) << 3);
            float v = c[t].e[r] * scale + (bias ? bias[m] : 0.0f);
            if (relu) v = v > 0.0f ? v : 0.0f;
            D[(size_t)m * dplane + col] = v;
        }
    }
}

// ------------------------- 3x3 conv via 9-tap WMMA -------------------------
// in: Cin x H x W (channel-major); wgt: [OC][Cin][3][3]; pad=1, stride param.
// Each wave: 16 oc x 64 output pixels (one row), 4 accumulators.
__global__ void conv3x3_wmma(const float* __restrict__ in, int H, int W,
                             const float* __restrict__ wgt,
                             const float* __restrict__ bias,
                             const float* __restrict__ bnsc,
                             const float* __restrict__ bnsh, int relu,
                             float* __restrict__ out, int OH, int OW,
                             int Cin, int stride, int xtiles)
{
    const int l   = threadIdx.x;
    const int oy  = blockIdx.x / xtiles;
    const int x0  = (blockIdx.x % xtiles) * 64;
    const int oc0 = blockIdx.y * 16;
    const int mA   = oc0 + (l & 15);
    const int khiA = (l >> 4) * 8;

    AHalf a, b[4];
    CAcc  c[4];
#pragma unroll
    for (int t = 0; t < 4; ++t)
#pragma unroll
        for (int r = 0; r < 8; ++r) c[t].e[r] = 0.0f;

    for (int tap = 0; tap < 9; ++tap) {
        const int dy = tap / 3, dx = tap % 3;
        const int iy = oy * stride + dy - 1;
        const bool rowok = (iy >= 0) && (iy < H);
        for (int c0 = 0; c0 < Cin; c0 += 32) {
#pragma unroll
            for (int h = 0; h < 16; ++h) {
                int kk = c0 + (h & 7) + ((h >> 3) << 4) + khiA;
                a.e[h] = (_Float16)wgt[((size_t)mA * Cin + kk) * 9 + tap];
            }
            const size_t rowoff = (size_t)(rowok ? iy : 0) * W;
#pragma unroll
            for (int t = 0; t < 4; ++t)
#pragma unroll
                for (int h = 0; h < 16; ++h) {
                    int cb = c0 + (h & 7) + ((h >> 3) << 4) + khiA;
                    size_t base = (size_t)cb * H * W + rowoff;
                    int ox = x0 + t * 16 + (l & 15);
                    int ix = ox * stride + dx - 1;
                    float v = 0.0f;
                    if (rowok && ox < OW && ix >= 0 && ix < W) v = in[base + ix];
                    b[t].e[h] = (_Float16)v;
                }
#pragma unroll
            for (int t = 0; t < 4; ++t)
                c[t].v = __builtin_amdgcn_wmma_f32_16x16x32_f16(false, a.v, false, b[t].v,
                                                                (short)0, c[t].v, false, false);
        }
    }

#pragma unroll
    for (int t = 0; t < 4; ++t) {
        const int ox = x0 + t * 16 + (l & 15);
        if (ox < OW) {
#pragma unroll
            for (int r = 0; r < 8; ++r) {
                int   m = oc0 + r + ((l >> 4) << 3);
                float v = c[t].e[r];
                if (bnsc) v = v * bnsc[m] + bnsh[m];
                else      v = v + (bias ? bias[m] : 0.0f);
                if (relu) v = v > 0.0f ? v : 0.0f;
                out[(size_t)m * OH * OW + (size_t)oy * OW + ox] = v;
            }
        }
    }
}

// --------------------------- small helper kernels --------------------------
__global__ void transpose_k(const float* __restrict__ in, float* __restrict__ o,
                            int N, int C)  // in: NxC -> out: CxN
{
    int idx = blockIdx.x * blockDim.x + threadIdx.x;
    if (idx >= N * C) return;
    int n = idx / C, cc = idx % C;
    o[(size_t)cc * N + n] = in[idx];
}

__global__ void bilinear_ac_k(const float* __restrict__ in, int C, int H, int W,
                              float* __restrict__ out, int OH, int OW)
{
    int idx = blockIdx.x * blockDim.x + threadIdx.x;
    if (idx >= C * OH * OW) return;
    int ox = idx % OW; int t = idx / OW; int oy = t % OH; int cc = t / OH;
    float fy = (OH > 1) ? (float)oy * (float)(H - 1) / (float)(OH - 1) : 0.0f;
    float fx = (OW > 1) ? (float)ox * (float)(W - 1) / (float)(OW - 1) : 0.0f;
    int y0 = (int)floorf(fy); y0 = y0 < 0 ? 0 : (y0 > H - 1 ? H - 1 : y0);
    int y1 = y0 + 1 > H - 1 ? H - 1 : y0 + 1;
    int x0 = (int)floorf(fx); x0 = x0 < 0 ? 0 : (x0 > W - 1 ? W - 1 : x0);
    int x1 = x0 + 1 > W - 1 ? W - 1 : x0 + 1;
    float wy = fy - (float)y0, wx = fx - (float)x0;
    const float* p = in + (size_t)cc * H * W;
    float r0 = p[y0 * W + x0] * (1.0f - wx) + p[y0 * W + x1] * wx;
    float r1 = p[y1 * W + x0] * (1.0f - wx) + p[y1 * W + x1] * wx;
    out[(size_t)cc * OH * OW + (size_t)oy * OW + ox] = r0 * (1.0f - wy) + r1 * wy;
}

__global__ void bn_fold_k(const float* g, const float* b, const float* m,
                          const float* v, const float* convb,
                          float* sc, float* sh, int n)
{
    int i = blockIdx.x * blockDim.x + threadIdx.x;
    if (i >= n) return;
    float inv = rsqrtf(v[i] + 1e-5f);
    float s = g[i] * inv;
    sc[i] = s;
    sh[i] = (convb[i] - m[i]) * s + b[i];
}

__global__ void softmax128_k(float* __restrict__ a)
{
    int p = blockIdx.x * blockDim.x + threadIdx.x;
    if (p >= HW336) return;
    float mx = -1e30f;
    for (int c = 0; c < 128; ++c) { float v = a[(size_t)c * HW336 + p]; mx = v > mx ? v : mx; }
    float s = 0.0f;
    for (int c = 0; c < 128; ++c) {
        float e = expf(a[(size_t)c * HW336 + p] - mx);
        a[(size_t)c * HW336 + p] = e; s += e;
    }
    float inv = 1.0f / s;
    for (int c = 0; c < 128; ++c) a[(size_t)c * HW336 + p] *= inv;
}

// -------- fused per-prototype grouped heads (h1->h2->h3) per 16x16 tile ----
__global__ void heads_k(const float* __restrict__ acts,
                        const float* __restrict__ w1, const float* __restrict__ s1,
                        const float* __restrict__ h1sh,
                        const float* __restrict__ w2, const float* __restrict__ s2,
                        const float* __restrict__ h2sh,
                        const float* __restrict__ w3, const float* __restrict__ b3,
                        float* __restrict__ res)
{
    const int p = blockIdx.y;            // prototype group 0..127
    const int ty = blockIdx.x / 21, tx = blockIdx.x % 21;
    const int y0 = ty * 16 - 3, x0 = tx * 16 - 3;   // 22x22 acts tile origin
    __shared__ float sA[22 * 22];
    __shared__ float sH1[8 * 20 * 20];
    __shared__ float sH2[8 * 18 * 18];
    const float* ap = acts + (size_t)p * HW336;

    for (int i = threadIdx.x; i < 22 * 22; i += blockDim.x) {
        int yy = y0 + i / 22, xx = x0 + i % 22;
        sA[i] = (yy >= 0 && yy < W336 && xx >= 0 && xx < W336) ? ap[yy * W336 + xx] : 0.0f;
    }
    __syncthreads();
    for (int i = threadIdx.x; i < 8 * 20 * 20; i += blockDim.x) {
        int o = i / 400, r = i % 400, hy = r / 20, hx = r % 20;
        const float* wp = w1 + (size_t)(p * 8 + o) * 9;
        float acc = 0.0f;
        for (int dy = 0; dy < 3; ++dy)
            for (int dx = 0; dx < 3; ++dx)
                acc += wp[dy * 3 + dx] * sA[(hy + dy) * 22 + hx + dx];
        int ch = p * 8 + o;
        acc = acc * s1[ch] + h1sh[ch];
        sH1[i] = acc > 0.0f ? acc : 0.0f;
    }
    __syncthreads();
    for (int i = threadIdx.x; i < 8 * 18 * 18; i += blockDim.x) {
        int o = i / 324, r = i % 324, hy = r / 18, hx = r % 18;
        float acc = 0.0f;
        for (int ci = 0; ci < 8; ++ci) {
            const float* wp = w2 + (size_t)((p * 8 + o) * 8 + ci) * 9;
            const float* hp = sH1 + ci * 400;
            for (int dy = 0; dy < 3; ++dy)
                for (int dx = 0; dx < 3; ++dx)
                    acc += wp[dy * 3 + dx] * hp[(hy + dy) * 20 + hx + dx];
        }
        int ch = p * 8 + o;
        acc = acc * s2[ch] + h2sh[ch];
        sH2[i] = acc > 0.0f ? acc : 0.0f;
    }
    __syncthreads();
    for (int i = threadIdx.x; i < 256; i += blockDim.x) {
        int hy = i / 16, hx = i % 16;
        float acc = b3[p];
        for (int ci = 0; ci < 8; ++ci) {
            const float* wp = w3 + (size_t)(p * 8 + ci) * 9;
            const float* hp = sH2 + ci * 324;
            for (int dy = 0; dy < 3; ++dy)
                for (int dx = 0; dx < 3; ++dx)
                    acc += wp[dy * 3 + dx] * hp[(hy + dy) * 18 + hx + dx];
        }
        res[(size_t)p * HW336 + (size_t)(ty * 16 + hy) * W336 + tx * 16 + hx] = tanhf(acc);
    }
}

__global__ void combine_k(const float* __restrict__ acts, const float* __restrict__ res,
                          const float* __restrict__ proto_bias,
                          const float* __restrict__ bias_scale,
                          const float* __restrict__ res_scale,
                          float* __restrict__ inst, float* __restrict__ wgt_o,
                          float* __restrict__ base_o)
{
    int p = blockIdx.x * blockDim.x + threadIdx.x;
    if (p >= HW336) return;
    float bs = bias_scale[0], rs = res_scale[0];
    float bb = 0.0f, ws = 0.0f;
    for (int c = 0; c < 128; ++c) {
        float a = acts[(size_t)c * HW336 + p];
        bb += a * proto_bias[c] * bs;
        ws += a * res[(size_t)c * HW336 + p];
    }
    wgt_o[p] = ws * rs;
    inst[p]  = bb + ws * rs;
    if (p < 128) base_o[p] = proto_bias[p] * bs;
}

// ---------------------------------------------------------------------------
extern "C" void kernel_launch(void* const* d_in, const int* in_sizes, int n_in,
                              void* d_out, int out_size, void* d_ws, size_t ws_size,
                              hipStream_t stream)
{
    (void)in_sizes; (void)n_in; (void)out_size; (void)ws_size;
    const int OC[4] = {256, 512, 1024, 1024};
    const int N576 = 576, DIRECT = 1 << 30;

    const float* X[4]  = {(const float*)d_in[0], (const float*)d_in[1],
                          (const float*)d_in[2], (const float*)d_in[3]};
    const float* projw[4] = {(const float*)d_in[4], (const float*)d_in[5],
                             (const float*)d_in[6], (const float*)d_in[7]};
    const float* projb[4] = {(const float*)d_in[8], (const float*)d_in[9],
                             (const float*)d_in[10], (const float*)d_in[11]};
    const float* rs0w = (const float*)d_in[12]; const float* rs0b = (const float*)d_in[13];
    const float* rs1w = (const float*)d_in[14]; const float* rs1b = (const float*)d_in[15];
    const float* rs3w = (const float*)d_in[16]; const float* rs3b = (const float*)d_in[17];
    const float* fusew = (const float*)d_in[18]; const float* fuseb = (const float*)d_in[19];
    const float* pe1w = (const float*)d_in[20]; const float* pe1b = (const float*)d_in[21];
    const float* pebn_g = (const float*)d_in[22]; const float* pebn_b = (const float*)d_in[23];
    const float* pebn_m = (const float*)d_in[24]; const float* pebn_v = (const float*)d_in[25];
    const float* pe2w = (const float*)d_in[26]; const float* pe2b = (const float*)d_in[27];
    const float* protos = (const float*)d_in[28];
    const float* proto_bias = (const float*)d_in[29];
    const float* bias_scale = (const float*)d_in[30];
    const float* res_scale  = (const float*)d_in[31];
    const float* smw = (const float*)d_in[32]; const float* smb = (const float*)d_in[33];
    const float* h1w = (const float*)d_in[34]; const float* h1b = (const float*)d_in[35];
    const float* bn1g = (const float*)d_in[36]; const float* bn1b = (const float*)d_in[37];
    const float* bn1m = (const float*)d_in[38]; const float* bn1v = (const float*)d_in[39];
    const float* h2w = (const float*)d_in[40]; const float* h2b = (const float*)d_in[41];
    const float* bn2g = (const float*)d_in[42]; const float* bn2b = (const float*)d_in[43];
    const float* bn2m = (const float*)d_in[44]; const float* bn2v = (const float*)d_in[45];
    const float* h3w = (const float*)d_in[46]; const float* h3b = (const float*)d_in[47];

    // workspace arena (byte offsets; fused region is reused by path336)
    const size_t MB = (size_t)1 << 20;
    char* ws = (char*)d_ws;
    float* fused   = (float*)(ws + 0);          // 2816*9216  (103.8MB)
    float* path336 = (float*)(ws + 0);          // 256*112896 (115.6MB, after fused dead)
    float* path96  = (float*)(ws + 116 * MB);   // 256*9216
    float* xT      = (float*)(ws + 126 * MB);   // 4 * 768*576
    float* fbuf    = (float*)(ws + 134 * MB);   // 2816*576 (f0..f3 planes)
    float* rs1buf  = (float*)(ws + 141 * MB);   // 512*2304
    float* rs3buf  = (float*)(ws + 146 * MB);   // 1024*144
    float* e1      = (float*)(ws + 147 * MB);   // 128*112896
    float* simbuf  = (float*)(ws + 205 * MB);   // 128*112896
    float* resbuf  = (float*)(ws + 263 * MB);   // 128*112896
    float* scPE    = (float*)(ws + 321 * MB);
    float* shPE = scPE + 1024; float* sc1 = scPE + 2048; float* sh1 = scPE + 3072;
    float* sc2 = scPE + 4096; float* sh2 = scPE + 5120;

    float* out    = (float*)d_out;
    float* inst_o = out;
    float* acts_o = out + HW336;
    float* base_o = out + HW336 + 128 * (size_t)HW336;
    float* wgt_o  = base_o + 128;
    float* emb_o  = wgt_o + HW336;

    auto gemm = [&](const float* A, int sAm, int sAk, const float* B, int ldb,
                    const float* bias, float scale, int relu, float* D, int dplane,
                    int M, int N, int K, int inW, int outW, int st, int ao, int bo) {
        dim3 g(N / 64, M / 16);
        gemm16_wmma<<<g, 32, 0, stream>>>(A, sAm, sAk, B, ldb, bias, scale, relu,
                                          D, dplane, K, inW, outW, st, ao, bo);
    };

    // 0) BN folds
    bn_fold_k<<<1, 128, 0, stream>>>(pebn_g, pebn_b, pebn_m, pebn_v, pe1b, scPE, shPE, 128);
    bn_fold_k<<<4, 256, 0, stream>>>(bn1g, bn1b, bn1m, bn1v, h1b, sc1, sh1, 1024);
    bn_fold_k<<<4, 256, 0, stream>>>(bn2g, bn2b, bn2m, bn2v, h2b, sc2, sh2, 1024);

    // 1) transpose tokens (N,C)->(C,N) so activations are channel-major
    for (int i = 0; i < 4; ++i)
        transpose_k<<<(576 * 768 + 255) / 256, 256, 0, stream>>>(X[i], xT + (size_t)i * 768 * 576, 576, 768);

    // 2) 1x1 projections (WMMA GEMM)
    float* fpl[4];
    {
        size_t off = 0;
        for (int i = 0; i < 4; ++i) { fpl[i] = fbuf + off; off += (size_t)OC[i] * N576; }
    }
    for (int i = 0; i < 4; ++i)
        gemm(projw[i], 768, 1, xT + (size_t)i * 768 * 576, N576, projb[i], 1.0f, 0,
             fpl[i], N576, OC[i], N576, 768, DIRECT, 0, 1, 0, 0);

    // 3) ConvTranspose rs0 (k=4,s=4): 16 sub-GEMMs scattered into fused[0:256] @96x96
    for (int a = 0; a < 4; ++a)
        for (int b = 0; b < 4; ++b)
            gemm(rs0w + (a * 4 + b), 16, 256 * 16, fpl[0], N576, rs0b, 1.0f, 0,
                 fused, HW96, 256, N576, 256, 24, 96, 4, a, b);

    // 4) ConvTranspose rs1 (k=2,s=2): 4 sub-GEMMs -> 512 x 48x48
    for (int a = 0; a < 2; ++a)
        for (int b = 0; b < 2; ++b)
            gemm(rs1w + (a * 2 + b), 4, 512 * 4, fpl[1], N576, rs1b, 1.0f, 0,
                 rs1buf, 2304, 512, N576, 512, 24, 48, 2, a, b);

    // 5) rs3: 3x3 stride-2 conv 1024->1024, 24x24 -> 12x12 (WMMA, 9 taps)
    conv3x3_wmma<<<dim3(1 * 12, 64), 32, 0, stream>>>(fpl[3], 24, 24, rs3w, rs3b,
                                                      nullptr, nullptr, 0,
                                                      rs3buf, 12, 12, 1024, 2, 1);

    // 6) bilinear-align-corners resizes into fused (channel slices)
    bilinear_ac_k<<<(512 * HW96 + 255) / 256, 256, 0, stream>>>(rs1buf, 512, 48, 48,
                                                                fused + (size_t)256 * HW96, 96, 96);
    bilinear_ac_k<<<(1024 * HW96 + 255) / 256, 256, 0, stream>>>(fpl[2], 1024, 24, 24,
                                                                 fused + (size_t)768 * HW96, 96, 96);
    bilinear_ac_k<<<(1024 * HW96 + 255) / 256, 256, 0, stream>>>(rs3buf, 1024, 12, 12,
                                                                 fused + (size_t)1792 * HW96, 96, 96);

    // 7) fuse: 1x1 conv 2816->256 + ReLU (WMMA GEMM)
    gemm(fusew, 2816, 1, fused, HW96, fuseb, 1.0f, 1, path96, HW96,
         256, HW96, 2816, DIRECT, 0, 1, 0, 0);

    // 8) upsample 96 -> 336 (fused region is dead, reuse as path336)
    bilinear_ac_k<<<((size_t)256 * HW336 + 255) / 256, 256, 0, stream>>>(path96, 256, 96, 96,
                                                                         path336, W336, W336);

    // 9) pe1: 3x3 conv 256->128 + BN + ReLU (WMMA)
    conv3x3_wmma<<<dim3(6 * W336, 8), 32, 0, stream>>>(path336, W336, W336, pe1w, nullptr,
                                                       scPE, shPE, 1,
                                                       e1, W336, W336, 256, 1, 6);
    // 10) pe2: 3x3 conv 128->64 -> emb (directly into d_out)
    conv3x3_wmma<<<dim3(6 * W336, 4), 32, 0, stream>>>(e1, W336, W336, pe2w, pe2b,
                                                       nullptr, nullptr, 0,
                                                       emb_o, W336, W336, 128, 1, 6);
    // 11) prototype einsum: sim = protos(128x64) x emb(64xHW) / 8 (WMMA GEMM)
    gemm(protos, 64, 1, emb_o, HW336, nullptr, 0.125f, 0, simbuf, HW336,
         128, HW336, 64, DIRECT, 0, 1, 0, 0);
    // 12) smooth: 3x3 conv 128->128 -> acts slice of d_out (pre-softmax)
    conv3x3_wmma<<<dim3(6 * W336, 8), 32, 0, stream>>>(simbuf, W336, W336, smw, smb,
                                                       nullptr, nullptr, 0,
                                                       acts_o, W336, W336, 128, 1, 6);
    // 13) softmax over 128 prototypes, in place
    softmax128_k<<<(HW336 + 255) / 256, 256, 0, stream>>>(acts_o);

    // 14) fused grouped heads h1->h2->h3 (LDS-tiled), res = tanh(...)
    heads_k<<<dim3(21 * 21, 128), 256, 0, stream>>>(acts_o, h1w, sc1, sh1,
                                                    h2w, sc2, sh2, h3w, h3b, resbuf);

    // 15) final reductions: inst / weighted / base broadcast
    combine_k<<<(HW336 + 255) / 256, 256, 0, stream>>>(acts_o, resbuf, proto_bias,
                                                       bias_scale, res_scale,
                                                       inst_o, wgt_o, base_o);
}
